// GATEmbeddings_51788715655450
// MI455X (gfx1250) — matrix operand order repaired and checked by reading, not compiled, not run on hardware
//
#include <hip/hip_runtime.h>
#include <hip/hip_bf16.h>

typedef __attribute__((ext_vector_type(16))) _Float16 v16h;
typedef __attribute__((ext_vector_type(8)))  float    v8f;

#define HEADS 8
#define DHID  64
#define HD    512   // HEADS*DHID
#define NEG_SLOPE 0.2f

// ---------------- feat gather: feat[n,:] = W[node_ids[n],:] ----------------
// 1 block (128 threads) per node, float4 copies (512 floats per row).
__global__ void k_gather_feat(const int* __restrict__ node_ids,
                              const float* __restrict__ W,
                              float* __restrict__ feat) {
    int n = blockIdx.x;
    long row = node_ids[n];
    const float4* s = (const float4*)(W + row * (long)HD);
    float4* d = (float4*)(feat + (long)n * HD);
    d[threadIdx.x] = s[threadIdx.x];
}

// ---------------- el/er via WMMA f32<-f16 16x16x32 ----------------
// grid = N/16 tiles, block = 256 (8 waves); wave w handles head w.
// A (16x32 f16) per ISA 16-bit A layout:
//   VGPR v in 0..3 : K = koff + 2v + t          (koff = (L>=16)?8:0)
//   VGPR v in 4..7 : K = 16 + koff + 2(v-4) + t
// B (32x16 f16): lane L covers column N=L%16, same K order as A:
//   element i of v16h = B[i<8 ? koff+i : 16+koff+i-8][N].  Col 0 = attn_l, col 1 = attn_r, rest 0.
//   Built branch-free: unconditional contiguous loads + FMA lane-mask select.
// D (16x16 f32): VGPR r -> M = r + (L>=16 ? 8 : 0), N = L%16.
__global__ void k_el_er_wmma(const float* __restrict__ feat,
                             const float* __restrict__ attn_l,
                             const float* __restrict__ attn_r,
                             float* __restrict__ el,
                             float* __restrict__ er) {
    int tile = blockIdx.x;
    int head = threadIdx.x >> 5;   // wave id = head
    int lane = threadIdx.x & 31;
    int M    = lane & 15;
    int node = tile * 16 + M;
    int koff = (lane >> 4) * 8;
    int Kh   = (lane >> 4) * 16;
    int N    = lane & 15;
    const float* fp  = feat   + (long)node * HD + head * DHID;
    const float* alp = attn_l + head * DHID;
    const float* arp = attn_r + head * DHID;
    const float w0 = (N == 0) ? 1.0f : 0.0f;
    const float w1 = (N == 1) ? 1.0f : 0.0f;

    v8f acc = {};
    #pragma unroll
    for (int kb = 0; kb < 2; ++kb) {           // two K=32 slabs cover D=64
        int kbase = kb * 32;
        v16h A, B;
        #pragma unroll
        for (int v = 0; v < 8; ++v) {
            int k = kbase + ((v < 4) ? (koff + 2 * v) : (16 + koff + 2 * (v - 4)));
            A[2 * v]     = (_Float16)fp[k];
            A[2 * v + 1] = (_Float16)fp[k + 1];
        }
        #pragma unroll
        for (int i = 0; i < 16; ++i) {
            int k = kbase + ((i < 8) ? (koff + i) : (16 + koff + (i - 8)));
            float bv = fmaf(w0, alp[k], w1 * arp[k]);   // unconditional loads
            B[i] = (_Float16)bv;
        }
        acc = __builtin_amdgcn_wmma_f32_16x16x32_f16(
                  false, A, false, B, (short)0, acc, false, false);
    }
    // Columns 0/1 of D hold el/er for the 16 tile rows.
    if (N <= 1) {
        int mbase = (lane >> 4) * 8;
        #pragma unroll
        for (int r = 0; r < 8; ++r) {
            int nd = tile * 16 + mbase + r;
            if (N == 0) el[nd * HEADS + head] = acc[r];
            else        er[nd * HEADS + head] = acc[r];
        }
    }
}

// ---------------- edge pass 1: e = leakyrelu(el[src]+er[dst]); segment max --
__device__ __forceinline__ unsigned f2key(float f) {
    unsigned b = __float_as_uint(f);
    return (b & 0x80000000u) ? ~b : (b | 0x80000000u);
}
__device__ __forceinline__ float key2f(unsigned k) {
    unsigned b = (k & 0x80000000u) ? (k ^ 0x80000000u) : ~k;
    return __uint_as_float(b);
}

__global__ void k_edge_logits(const int* __restrict__ src,
                              const int* __restrict__ dst,
                              const float* __restrict__ el,
                              const float* __restrict__ er,
                              float* __restrict__ ebuf,
                              unsigned* __restrict__ mkey,
                              int total) {
    int idx = blockIdx.x * blockDim.x + threadIdx.x;
    if (idx >= total) return;
    int e = idx >> 3, h = idx & 7;
    float v = el[src[e] * HEADS + h] + er[dst[e] * HEADS + h];
    v = (v > 0.0f) ? v : NEG_SLOPE * v;
    ebuf[idx] = v;
    atomicMax(&mkey[dst[e] * HEADS + h], f2key(v));
}

// ---------------- edge pass 2: p = exp(e - m[dst]); segment sum ------------
__global__ void k_edge_exp(const int* __restrict__ dst,
                           float* __restrict__ ebuf,
                           const unsigned* __restrict__ mkey,
                           float* __restrict__ denom,
                           int total) {
    int idx = blockIdx.x * blockDim.x + threadIdx.x;
    if (idx >= total) return;
    int e = idx >> 3, h = idx & 7;
    float m = key2f(mkey[dst[e] * HEADS + h]);
    float p = __expf(ebuf[idx] - m);
    ebuf[idx] = p;
    atomicAdd(&denom[dst[e] * HEADS + h], p);
}

// ---------------- edge pass 3: accum[dst] += (p/denom[dst]) * feat[src] ----
// 2 blocks of 256 threads per edge; e derived from blockIdx so src/dst/ebuf/
// denom accesses are wave-uniform (scalar path), only feat-gather + f32
// atomic-add stay in the vector path.  feat & accum are L2-resident (32MB ea).
__global__ void k_aggregate(const int* __restrict__ src,
                            const int* __restrict__ dst,
                            const float* __restrict__ ebuf,
                            const float* __restrict__ denom,
                            const float* __restrict__ feat,
                            float* __restrict__ accum) {
    int e    = blockIdx.x >> 1;
    int hd   = ((blockIdx.x & 1) << 8) + threadIdx.x;   // 0..511
    int h    = hd >> 6;
    int s = src[e], d = dst[e];
    float a = ebuf[e * HEADS + h] / denom[d * HEADS + h];
    atomicAdd(&accum[(long)d * HD + hd], a * feat[(long)s * HD + hd]);
}

// ---------------- ELU epilogue into d_out ----------------------------------
__global__ void k_elu(const float* __restrict__ accum,
                      float* __restrict__ out, int total) {
    int idx = blockIdx.x * blockDim.x + threadIdx.x;
    if (idx >= total) return;
    float x = accum[idx];
    out[idx] = (x > 0.0f) ? x : (__expf(x) - 1.0f);
}

extern "C" void kernel_launch(void* const* d_in, const int* in_sizes, int n_in,
                              void* d_out, int out_size, void* d_ws, size_t ws_size,
                              hipStream_t stream) {
    const int*   node_ids = (const int*)  d_in[0];
    const int*   src      = (const int*)  d_in[1];
    const int*   dst      = (const int*)  d_in[2];
    const float* W        = (const float*)d_in[3];
    const float* attn_l   = (const float*)d_in[4];
    const float* attn_r   = (const float*)d_in[5];
    float* out = (float*)d_out;

    const int N = in_sizes[0];            // 16384
    const int E = in_sizes[1];            // 524288

    // workspace layout (4-byte units); accum+denom+mkey contiguous for one
    // memset.
    float* ws = (float*)d_ws;
    float*    feat  = ws;                               // N*HD
    float*    accum = feat  + (size_t)N * HD;           // N*HD
    float*    denom = accum + (size_t)N * HD;           // N*HEADS
    unsigned* mkey  = (unsigned*)(denom + (size_t)N * HEADS);   // N*HEADS
    float*    el    = (float*)(mkey + (size_t)N * HEADS);       // N*HEADS
    float*    er    = el + (size_t)N * HEADS;                   // N*HEADS
    float*    ebuf  = er + (size_t)N * HEADS;                   // E*HEADS

    // zero accum, denom, mkey (mkey==0 is only read for (dst,h) slots that
    // pass 1 has already maxed with a real key > 0).
    hipMemsetAsync(accum, 0, ((size_t)N * HD + 2u * N * HEADS) * sizeof(float),
                   stream);

    k_gather_feat<<<N, 128, 0, stream>>>(node_ids, W, feat);

    k_el_er_wmma<<<N / 16, 256, 0, stream>>>(feat, attn_l, attn_r, el, er);

    int totEH = E * HEADS;
    k_edge_logits<<<(totEH + 255) / 256, 256, 0, stream>>>(src, dst, el, er,
                                                           ebuf, mkey, totEH);
    k_edge_exp<<<(totEH + 255) / 256, 256, 0, stream>>>(dst, ebuf, mkey,
                                                        denom, totEH);

    k_aggregate<<<E * 2, 256, 0, stream>>>(src, dst, ebuf, denom, feat, accum);

    int totOut = N * HD;
    k_elu<<<(totOut + 255) / 256, 256, 0, stream>>>(accum, out, totOut);
}
